// Translate_13795434955464
// MI455X (gfx1250) — compile-verified, matrix operand order checked by reading, not run on hardware
//
#include <hip/hip_runtime.h>
#include <hip/hip_bf16.h>

// ---------------------------------------------------------------------------
// MI455X (gfx1250) bf16-WMMA pipeline for the dual-autoencoder loss.
//   H1 = X1@W1h + b ; H2 = X2@W2h + b        (16384x1024 @ 1024x2048)
//   L1 = rowsq(H1@Wh2 + b - X2) ; L2 = rowsq(H2@Wh1 + b - X1)
//   L3 = rowsq(H1-H2) ; L4 = 0.5*((H1+H2)@who + b) - Y
//   loss = sum(L1^2+L2^2+L3^2+L4^2)
// GEMMs: v_wmma_f32_16x16x32_bf16, wave32, 128x128 tiles,
// double-buffered LDS fed by global_load_async_to_lds_b128 (ASYNCcnt),
// software-pipelined (steady-state always prefetches; last tile peeled).
// ---------------------------------------------------------------------------

typedef __attribute__((ext_vector_type(16))) __bf16 v16bf;
typedef __attribute__((ext_vector_type(8)))  __bf16 v8bf;
typedef __attribute__((ext_vector_type(8)))  float  v8f;

#define NROWS 16384
#define DXDIM 1024
#define DHDIM 2048
#define LDSS  40                    // padded LDS row stride (halfwords)
#define TILEHW (128 * LDSS)         // halfwords per tile buffer
#define TILEB  (TILEHW * 2)         // bytes per tile buffer (10240)

__device__ __forceinline__ unsigned short f2bf(float f) {
  unsigned u = __float_as_uint(f);
  u += 0x7FFFu + ((u >> 16) & 1u);        // round-to-nearest-even
  return (unsigned short)(u >> 16);
}
__device__ __forceinline__ float bf2f(unsigned short h) {
  return __uint_as_float(((unsigned)h) << 16);
}

// CDNA5 async global->LDS copy (bypasses VGPRs, tracked by ASYNCcnt).
__device__ __forceinline__ void async_ld_b128(unsigned lds_addr, const void* gaddr) {
  asm volatile("global_load_async_to_lds_b128 %0, %1, off"
               :: "v"(lds_addr), "v"(gaddr) : "memory");
}
__device__ __forceinline__ void wait_async4() {
  asm volatile("s_wait_asynccnt 0x4" ::: "memory");
}
__device__ __forceinline__ void wait_async0() {
  asm volatile("s_wait_asynccnt 0x0" ::: "memory");
}

// ---------------- elementwise converts -------------------------------------
__global__ __launch_bounds__(256)
void cvt_bf16_kernel(const float* __restrict__ in, unsigned short* __restrict__ out, int n) {
  int i = (blockIdx.x * blockDim.x + threadIdx.x) * 4;
  if (i < n) {
    float4 v = *(const float4*)(in + i);
    ushort4 o;
    o.x = f2bf(v.x); o.y = f2bf(v.y); o.z = f2bf(v.z); o.w = f2bf(v.w);
    *(ushort4*)(out + i) = o;
  }
}

// W[din][dout] f32 -> WT[dout][din] bf16 (so GEMM B-operand is K-contiguous)
__global__ __launch_bounds__(256)
void cvt_wT_kernel(const float* __restrict__ in, unsigned short* __restrict__ out,
                   int din, int dout) {
  int idx = blockIdx.x * blockDim.x + threadIdx.x;
  if (idx >= din * dout) return;
  int k = idx / dout, n = idx % dout;
  out[(size_t)n * din + k] = f2bf(in[idx]);
}

__global__ __launch_bounds__(256)
void zero_kernel(float* __restrict__ p, int n) {
  int i = blockIdx.x * blockDim.x + threadIdx.x;
  if (i < n) p[i] = 0.0f;
}

// ---------------- fragment helper ------------------------------------------
__device__ __forceinline__ v16bf load_frag(const unsigned short* __restrict__ s,
                                           int row, int kb) {
  union { v16bf v; v8bf h[2]; } u;
  u.h[0] = *(const v8bf*)&s[row * LDSS + kb];        // K = kb..kb+7
  u.h[1] = *(const v8bf*)&s[row * LDSS + kb + 16];   // K = kb+16..kb+23
  return u.v;
}

// Shared GEMM core: accumulates a 128x128 tile of A@B^T into acc[2][4].
// K_ is a compile-time constant (templated callers) so the pipeline fully
// strength-reduces; steady-state loop always prefetches, last tile peeled.
#define GEMM_CORE(A_, BT_, K_)                                                 \
  __shared__ unsigned short sMem[4 * TILEHW]; /* A0|A1|B0|B1, 40KB */          \
  const int tid  = threadIdx.x;                                                \
  const int lane = tid & 31;                                                   \
  const int wave = tid >> 5;                                                   \
  const int wm = wave & 3;                                                     \
  const int wn = wave >> 2;                                                    \
  const int hh = lane >> 4;                                                    \
  const int lr = lane & 15;                                                    \
  const int m0 = blockIdx.x * 128;                                             \
  const int n0 = blockIdx.y * 128;                                             \
  v8f acc[2][4];                                                               \
  _Pragma("unroll")                                                            \
  for (int mi = 0; mi < 2; ++mi)                                               \
    _Pragma("unroll")                                                          \
    for (int ni = 0; ni < 4; ++ni)                                             \
      acc[mi][ni] = (v8f){0.f,0.f,0.f,0.f,0.f,0.f,0.f,0.f};                    \
  /* per-thread staging addresses: rows (tid>>2, +64), 16B segment tid&3 */    \
  const int srow = tid >> 2, sseg = tid & 3;                                   \
  const unsigned short* gA0 = A_  + (size_t)(m0 + srow) * K_ + sseg * 8;       \
  const unsigned short* gA1 = gA0 + (size_t)64 * K_;                           \
  const unsigned short* gB0 = BT_ + (size_t)(n0 + srow) * K_ + sseg * 8;       \
  const unsigned short* gB1 = gB0 + (size_t)64 * K_;                           \
  const unsigned ldsBase = (unsigned)(uintptr_t)sMem;                          \
  const unsigned lA0 = ldsBase + (srow * LDSS + sseg * 8) * 2;                 \
  const unsigned lA1 = lA0 + 64 * LDSS * 2;                                    \
  const unsigned lB0 = lA0 + 2 * TILEB;                                        \
  const unsigned lB1 = lA1 + 2 * TILEB;                                        \
  constexpr int niter = (K_) / 32;                                             \
  /* prologue: stage tile 0 into buffer 0 */                                   \
  async_ld_b128(lA0, gA0); async_ld_b128(lA1, gA1);                            \
  async_ld_b128(lB0, gB0); async_ld_b128(lB1, gB1);                            \
  gA0 += 32; gA1 += 32; gB0 += 32; gB1 += 32;                                  \
  _Pragma("unroll 2")                                                          \
  for (int it = 0; it < niter - 1; ++it) {                                     \
    const unsigned nb = (unsigned)((it + 1) & 1) * TILEB;                      \
    async_ld_b128(lA0 + nb, gA0); async_ld_b128(lA1 + nb, gA1);                \
    async_ld_b128(lB0 + nb, gB0); async_ld_b128(lB1 + nb, gB1);                \
    gA0 += 32; gA1 += 32; gB0 += 32; gB1 += 32;                                \
    wait_async4();              /* tile 'it' landed, next 4 in flight */       \
    __syncthreads();            /* all waves' async writes visible */          \
    GEMM_TILE_MMA(it)                                                          \
    __syncthreads();            /* done reading before buffer is refilled */   \
  }                                                                            \
  /* epilogue: last tile, nothing in flight behind it */                       \
  wait_async0();                                                               \
  __syncthreads();                                                             \
  GEMM_TILE_MMA(niter - 1)

#define GEMM_TILE_MMA(IT_)                                                     \
  {                                                                            \
    const unsigned short* sAcur = sMem + ((IT_) & 1) * TILEHW;                 \
    const unsigned short* sBcur = sMem + 2 * TILEHW + ((IT_) & 1) * TILEHW;    \
    const int kb = hh * 8;                                                     \
    v16bf af[2], bfm[4];                                                       \
    _Pragma("unroll")                                                          \
    for (int mi = 0; mi < 2; ++mi) af[mi]  = load_frag(sAcur, wm*32+mi*16+lr, kb); \
    _Pragma("unroll")                                                          \
    for (int ni = 0; ni < 4; ++ni) bfm[ni] = load_frag(sBcur, wn*64+ni*16+lr, kb); \
    _Pragma("unroll")                                                          \
    for (int mi = 0; mi < 2; ++mi)                                             \
      _Pragma("unroll")                                                        \
      for (int ni = 0; ni < 4; ++ni)                                           \
        acc[mi][ni] = __builtin_amdgcn_wmma_f32_16x16x32_bf16(                 \
            false, af[mi], false, bfm[ni], (short)0, acc[mi][ni], false, false); \
  }

// ---------------- GEMM 1: H = bf16(A @ B^T + bias) -------------------------
template <int K_>
__global__ __launch_bounds__(256)
void gemm_h_kernel(const unsigned short* __restrict__ A,   // [M][K] bf16
                   const unsigned short* __restrict__ BT,  // [Nout][K] bf16
                   const float* __restrict__ bias,         // [Nout]
                   unsigned short* __restrict__ H,         // [M][Nout] bf16
                   int Nout) {
  GEMM_CORE(A, BT, K_)

#pragma unroll
  for (int ni = 0; ni < 4; ++ni) {
    int col = n0 + wn * 64 + ni * 16 + lr;
    float bc = bias[col];
#pragma unroll
    for (int mi = 0; mi < 2; ++mi) {
      int rbase = m0 + wm * 32 + mi * 16 + hh * 8;
#pragma unroll
      for (int e = 0; e < 8; ++e)
        H[(size_t)(rbase + e) * Nout + col] = f2bf(acc[mi][ni][e] + bc);
    }
  }
}

// ------- GEMM 2 + fused recon loss: L[row] += sum_col (A@B^T + b - X)^2 ----
template <int K_>
__global__ __launch_bounds__(256)
void gemm_recon_kernel(const unsigned short* __restrict__ A,   // [M][K] bf16 (H)
                       const unsigned short* __restrict__ BT,  // [Nout][K] bf16
                       const float* __restrict__ bias,         // [Nout]
                       const float* __restrict__ Xref,         // [M][Nout] f32
                       float* __restrict__ Lout,               // [M]
                       int Nout) {
  GEMM_CORE(A, BT, K_)

  // per-row squared-error partials (each lane owns 16 distinct rows x 4 cols)
  float rs[2][8];
#pragma unroll
  for (int mi = 0; mi < 2; ++mi)
#pragma unroll
    for (int e = 0; e < 8; ++e) rs[mi][e] = 0.0f;

#pragma unroll
  for (int ni = 0; ni < 4; ++ni) {
    int col = n0 + wn * 64 + ni * 16 + lr;
    float bc = bias[col];
#pragma unroll
    for (int mi = 0; mi < 2; ++mi) {
      int rbase = m0 + wm * 32 + mi * 16 + hh * 8;
#pragma unroll
      for (int e = 0; e < 8; ++e) {
        float d = acc[mi][ni][e] + bc - Xref[(size_t)(rbase + e) * Nout + col];
        rs[mi][e] += d * d;
      }
    }
  }
  // lanes 0..15 (and 16..31) share rows -> reduce across the half-wave
#pragma unroll
  for (int mi = 0; mi < 2; ++mi)
#pragma unroll
    for (int e = 0; e < 8; ++e) {
      float v = rs[mi][e];
      v += __shfl_xor(v, 1, 32);
      v += __shfl_xor(v, 2, 32);
      v += __shfl_xor(v, 4, 32);
      v += __shfl_xor(v, 8, 32);
      if (lr == 0)
        atomicAdd(&Lout[m0 + wm * 32 + mi * 16 + hh * 8 + e], v);
    }
}

// ---------------- L3 / L4 (one wave per row) --------------------------------
__global__ __launch_bounds__(256)
void l34_kernel(const unsigned short* __restrict__ H1,
                const unsigned short* __restrict__ H2,
                const float* __restrict__ whoW, const float* __restrict__ whob,
                const float* __restrict__ Y,
                float* __restrict__ L3, float* __restrict__ L4) {
  int wave = threadIdx.x >> 5, lane = threadIdx.x & 31;
  int row = blockIdx.x * 8 + wave;
  const unsigned short* h1 = H1 + (size_t)row * DHDIM;
  const unsigned short* h2 = H2 + (size_t)row * DHDIM;
  float s3 = 0.f, s4 = 0.f;
  for (int j = lane; j < DHDIM; j += 32) {
    float a = bf2f(h1[j]), b = bf2f(h2[j]);
    float d = a - b;
    s3 += d * d;
    s4 += (a + b) * whoW[j];
  }
  for (int m = 1; m < 32; m <<= 1) {
    s3 += __shfl_xor(s3, m, 32);
    s4 += __shfl_xor(s4, m, 32);
  }
  if (lane == 0) {
    L3[row] = s3;
    L4[row] = 0.5f * (s4 + whob[0]) - Y[row];
  }
}

// ---------------- final scalar: sum of squares of L1..L4 -------------------
__global__ __launch_bounds__(1024)
void final_kernel(const float* __restrict__ L, float* __restrict__ out) {
  __shared__ float red[32];
  float s = 0.f;
  for (int i = threadIdx.x; i < 4 * NROWS; i += 1024) {
    float v = L[i];
    s += v * v;
  }
  for (int m = 1; m < 32; m <<= 1) s += __shfl_xor(s, m, 32);
  int lane = threadIdx.x & 31, wave = threadIdx.x >> 5;
  if (lane == 0) red[wave] = s;
  __syncthreads();
  if (wave == 0) {
    float t = red[lane];
    for (int m = 1; m < 32; m <<= 1) t += __shfl_xor(t, m, 32);
    if (lane == 0) out[0] = t;
  }
}

// ---------------------------------------------------------------------------
extern "C" void kernel_launch(void* const* d_in, const int* in_sizes, int n_in,
                              void* d_out, int out_size, void* d_ws, size_t ws_size,
                              hipStream_t stream) {
  const float* X1   = (const float*)d_in[0];
  const float* X2   = (const float*)d_in[1];
  const float* Y    = (const float*)d_in[2];
  const float* w1hW = (const float*)d_in[3];
  const float* w1hb = (const float*)d_in[4];
  const float* w2hW = (const float*)d_in[5];
  const float* w2hb = (const float*)d_in[6];
  const float* wh1W = (const float*)d_in[7];
  const float* wh1b = (const float*)d_in[8];
  const float* wh2W = (const float*)d_in[9];
  const float* wh2b = (const float*)d_in[10];
  const float* whoW = (const float*)d_in[11];
  const float* whob = (const float*)d_in[12];

  char* ws = (char*)d_ws;
  unsigned short* X1b = (unsigned short*)(ws + 0);          // 32 MB
  unsigned short* X2b = (unsigned short*)(ws + 33554432);   // 32 MB
  unsigned short* W1T = (unsigned short*)(ws + 67108864);   // [DH][DX] 4 MB
  unsigned short* W2T = (unsigned short*)(ws + 71303168);   // 4 MB
  unsigned short* V1T = (unsigned short*)(ws + 75497472);   // wh1^T [DX][DH] 4 MB
  unsigned short* V2T = (unsigned short*)(ws + 79691776);   // 4 MB
  unsigned short* H1  = (unsigned short*)(ws + 83886080);   // 64 MB
  unsigned short* H2  = (unsigned short*)(ws + 150994944);  // 64 MB
  float* L1 = (float*)(ws + 218103808);                     // 4 x 64 KB
  float* L2 = L1 + NROWS;
  float* L3 = L2 + NROWS;
  float* L4 = L3 + NROWS;

  // stage 0: precision conversion + weight transposes (bandwidth-trivial)
  cvt_bf16_kernel<<<(NROWS * DXDIM / 4 + 255) / 256, 256, 0, stream>>>(X1, X1b, NROWS * DXDIM);
  cvt_bf16_kernel<<<(NROWS * DXDIM / 4 + 255) / 256, 256, 0, stream>>>(X2, X2b, NROWS * DXDIM);
  cvt_wT_kernel<<<(DXDIM * DHDIM + 255) / 256, 256, 0, stream>>>(w1hW, W1T, DXDIM, DHDIM);
  cvt_wT_kernel<<<(DXDIM * DHDIM + 255) / 256, 256, 0, stream>>>(w2hW, W2T, DXDIM, DHDIM);
  cvt_wT_kernel<<<(DXDIM * DHDIM + 255) / 256, 256, 0, stream>>>(wh1W, V1T, DHDIM, DXDIM);
  cvt_wT_kernel<<<(DXDIM * DHDIM + 255) / 256, 256, 0, stream>>>(wh2W, V2T, DHDIM, DXDIM);
  zero_kernel<<<(2 * NROWS + 255) / 256, 256, 0, stream>>>(L1, 2 * NROWS);

  // stage 1: H1, H2 (WMMA bf16, f32 accumulate), K = 1024
  dim3 g1(NROWS / 128, DHDIM / 128);
  gemm_h_kernel<DXDIM><<<g1, 256, 0, stream>>>(X1b, W1T, w1hb, H1, DHDIM);
  gemm_h_kernel<DXDIM><<<g1, 256, 0, stream>>>(X2b, W2T, w2hb, H2, DHDIM);

  // stage 2: reconstruction GEMMs with fused squared-error row sums, K = 2048
  dim3 g2(NROWS / 128, DXDIM / 128);
  gemm_recon_kernel<DHDIM><<<g2, 256, 0, stream>>>(H1, V2T, wh2b, X2, L1, DXDIM);
  gemm_recon_kernel<DHDIM><<<g2, 256, 0, stream>>>(H2, V1T, wh1b, X1, L2, DXDIM);

  // stage 3: L3/L4 and final scalar reduction
  l34_kernel<<<NROWS / 8, 256, 0, stream>>>(H1, H2, whoW, whob, Y, L3, L4);
  final_kernel<<<1, 1024, 0, stream>>>(L1, (float*)d_out);
}
